// HeteroMessagePassingBlock_28381143892393
// MI455X (gfx1250) — matrix operand order repaired and checked by reading, not compiled, not run on hardware
//
#include <hip/hip_runtime.h>
#include <cmath>

typedef __bf16 v16bf __attribute__((ext_vector_type(16)));
typedef float  v8f   __attribute__((ext_vector_type(8)));

constexpr int NN  = 50000;
constexpr int RR  = 6;
constexpr int EE  = 32768;
constexpr int HH  = 256;
constexpr int FF  = 16;
constexpr int NHH = 8;
constexpr int HD  = 32;
constexpr int KVP = 288;   // 272 padded to multiple of 32

// ---------------- scalar helpers ----------------
__device__ __forceinline__ __bf16 f2bf(float f) {
    unsigned u = __float_as_uint(f);
    u = (u + 0x7fffu + ((u >> 16) & 1u)) >> 16;    // round-to-nearest-even
    union { unsigned short s; __bf16 b; } cv;
    cv.s = (unsigned short)u;
    return cv.b;
}
__device__ __forceinline__ float bf2f(__bf16 b) {
    union { __bf16 b; unsigned short s; } cv;
    cv.b = b;
    return __uint_as_float(((unsigned)cv.s) << 16);
}
__device__ __forceinline__ float gelu_f(float v) {
    return 0.5f * v * (1.0f + erff(v * 0.70710678118654752f));
}
__device__ __forceinline__ float epi_apply(int EPI, float v) {
    if (EPI == 1) return gelu_f(v);
    if (EPI == 2) return tanhf(v);
    return v;
}
__device__ __forceinline__ void atomicMaxF(float* addr, float val) {
    unsigned* ua = (unsigned*)addr;
    unsigned old = *ua;
    while (true) {
        float f = __uint_as_float(old);
        if (f >= val) break;
        unsigned assumed = old;
        old = atomicCAS(ua, assumed, __float_as_uint(val));
        if (old == assumed) break;
    }
}

// ---------------- WMMA fragment loads ----------------
// A 16x32 bf16 (row-major source). p -> &A[row][k0] for this lane's row.
// lanes 0-15: K pairs {0,2,...,14}; lanes 16-31: +16.
__device__ __forceinline__ v16bf frag_a(const __bf16* p, int hi8) {
    union { v16bf v; unsigned u[8]; } f;
#pragma unroll
    for (int j = 0; j < 8; ++j) {
        int off = 2 * j + hi8 * 2;
        f.u[j] = *reinterpret_cast<const unsigned*>(p + off);
    }
    return f.v;
}
// B 32x16 bf16 loaded from transposed weights Bt[Ncols][Kpad].
// p -> &Bt[col][k0 + (lane>=16 ? 16 : 0)]; VGPR j holds K = +2j, +2j+1.
__device__ __forceinline__ v16bf frag_b(const __bf16* p) {
    union { v16bf v; unsigned u[8]; } f;
#pragma unroll
    for (int j = 0; j < 8; ++j)
        f.u[j] = *reinterpret_cast<const unsigned*>(p + 2 * j);
    return f.v;
}

// ---------------- big-tile GEMM: 32(M) x 64(N) macro-tile per wave ----------------
// C[M,Nc] = A[M,K] @ Bt^T + bias.  EPI: 0 none, 1 gelu, 2 tanh.
// Per K-step: 2 A frags + 4 B frags feed 8 wmma ops (1.5 b128 loads / wmma).
template <int EPI>
__global__ void gemm_big(const __bf16* __restrict__ A, long lda,
                         const __bf16* __restrict__ Bt, long ldb, int K,
                         const float* __restrict__ bias,
                         float* __restrict__ C, __bf16* __restrict__ Cb, long ldc,
                         int M, int tilesM, int tilesN, int nValid) {
    int wave = (int)((blockIdx.x * blockDim.x + threadIdx.x) >> 5);
    if (wave >= tilesM * tilesN) return;
    int lane = threadIdx.x & 31;
    int tm = wave / tilesN, tn = wave % tilesN;
    int lrow = lane & 15;
    int hi = (lane >> 4) & 1;
    int r0 = tm * 32 + lrow;      if (r0 > M - 1) r0 = M - 1;
    int r1 = tm * 32 + 16 + lrow; if (r1 > M - 1) r1 = M - 1;
    const __bf16* arow0 = A + (long)r0 * lda;
    const __bf16* arow1 = A + (long)r1 * lda;
    const __bf16* brow  = Bt + (long)(tn * 64 + lrow) * ldb + hi * 16;
    long bs16 = 16 * ldb;
    v8f acc[8];
#pragma unroll
    for (int i = 0; i < 8; ++i) acc[i] = {0.f, 0.f, 0.f, 0.f, 0.f, 0.f, 0.f, 0.f};
    for (int k0 = 0; k0 < K; k0 += 32) {
        v16bf a0 = frag_a(arow0 + k0, hi * 8);
        v16bf a1 = frag_a(arow1 + k0, hi * 8);
#pragma unroll
        for (int j = 0; j < 4; ++j) {
            v16bf b = frag_b(brow + j * bs16 + k0);
            acc[j]     = __builtin_amdgcn_wmma_f32_16x16x32_bf16(false, a0, false, b,
                                                                 (short)0, acc[j], false, false);
            acc[4 + j] = __builtin_amdgcn_wmma_f32_16x16x32_bf16(false, a1, false, b,
                                                                 (short)0, acc[4 + j], false, false);
        }
    }
#pragma unroll
    for (int j = 0; j < 4; ++j) {
        int col = tn * 64 + j * 16 + lrow;
        if (col >= nValid) continue;
        float bv = bias ? bias[col] : 0.f;
#pragma unroll
        for (int sm = 0; sm < 2; ++sm) {
#pragma unroll
            for (int i = 0; i < 8; ++i) {
                int row = tm * 32 + sm * 16 + i + hi * 8;
                if (row < M) {
                    float v = epi_apply(EPI, acc[sm * 4 + j][i] + bv);
                    if (C)  C[(long)row * ldc + col] = v;
                    if (Cb) Cb[(long)row * ldc + col] = f2bf(v);
                }
            }
        }
    }
}

// big-tile GEMM over A = rel_out[R][N][H] viewed as [N, R*H]
template <int EPI>
__global__ void gemm_relcat_big(const __bf16* __restrict__ rel,
                                const __bf16* __restrict__ Bt,
                                const float* __restrict__ bias,
                                float* __restrict__ C, __bf16* __restrict__ Cb, long ldc,
                                int tilesM, int tilesN, int nValid) {
    int wave = (int)((blockIdx.x * blockDim.x + threadIdx.x) >> 5);
    if (wave >= tilesM * tilesN) return;
    int lane = threadIdx.x & 31;
    int tm = wave / tilesN, tn = wave % tilesN;
    int lrow = lane & 15;
    int hi = (lane >> 4) & 1;
    int r0 = tm * 32 + lrow;      if (r0 > NN - 1) r0 = NN - 1;
    int r1 = tm * 32 + 16 + lrow; if (r1 > NN - 1) r1 = NN - 1;
    const __bf16* brow = Bt + (long)(tn * 64 + lrow) * (RR * HH) + hi * 16;
    long bs16 = 16 * (RR * HH);
    v8f acc[8];
#pragma unroll
    for (int i = 0; i < 8; ++i) acc[i] = {0.f, 0.f, 0.f, 0.f, 0.f, 0.f, 0.f, 0.f};
    for (int k0 = 0; k0 < RR * HH; k0 += 32) {
        int rr = k0 >> 8;
        const __bf16* abase = rel + (long)rr * NN * HH + (k0 & 255);
        v16bf a0 = frag_a(abase + (long)r0 * HH, hi * 8);
        v16bf a1 = frag_a(abase + (long)r1 * HH, hi * 8);
#pragma unroll
        for (int j = 0; j < 4; ++j) {
            v16bf b = frag_b(brow + j * bs16 + k0);
            acc[j]     = __builtin_amdgcn_wmma_f32_16x16x32_bf16(false, a0, false, b,
                                                                 (short)0, acc[j], false, false);
            acc[4 + j] = __builtin_amdgcn_wmma_f32_16x16x32_bf16(false, a1, false, b,
                                                                 (short)0, acc[4 + j], false, false);
        }
    }
#pragma unroll
    for (int j = 0; j < 4; ++j) {
        int col = tn * 64 + j * 16 + lrow;
        if (col >= nValid) continue;
        float bv = bias ? bias[col] : 0.f;
#pragma unroll
        for (int sm = 0; sm < 2; ++sm) {
#pragma unroll
            for (int i = 0; i < 8; ++i) {
                int row = tm * 32 + sm * 16 + i + hi * 8;
                if (row < NN) {
                    float v = epi_apply(EPI, acc[sm * 4 + j][i] + bv);
                    if (C)  C[(long)row * ldc + col] = v;
                    if (Cb) Cb[(long)row * ldc + col] = f2bf(v);
                }
            }
        }
    }
}

// small GEMM (one 16x16 tile per wave) -- used only for the tiny logits GEMM (Nc=16)
template <int EPI>
__global__ void gemm_bf16(const __bf16* __restrict__ A, long lda,
                          const __bf16* __restrict__ Bt, long ldb, int K,
                          const float* __restrict__ bias,
                          float* __restrict__ C, __bf16* __restrict__ Cb, long ldc,
                          int tilesM, int tilesN, int nValid) {
    int wave = (int)((blockIdx.x * blockDim.x + threadIdx.x) >> 5);
    if (wave >= tilesM * tilesN) return;
    int lane = threadIdx.x & 31;
    int tm = wave / tilesN, tn = wave % tilesN;
    int lrow = lane & 15;
    int hi = (lane >> 4) & 1;
    const __bf16* arow = A + (long)(tm * 16 + lrow) * lda;
    const __bf16* brow = Bt + (long)(tn * 16 + lrow) * ldb + hi * 16;
    v8f acc = {0.f, 0.f, 0.f, 0.f, 0.f, 0.f, 0.f, 0.f};
    for (int k0 = 0; k0 < K; k0 += 32) {
        v16bf a = frag_a(arow + k0, hi * 8);
        v16bf b = frag_b(brow + k0);
        acc = __builtin_amdgcn_wmma_f32_16x16x32_bf16(false, a, false, b,
                                                      (short)0, acc, false, false);
    }
    int col = tn * 16 + lrow;
    if (col < nValid) {
        float bv = bias ? bias[col] : 0.f;
#pragma unroll
        for (int i = 0; i < 8; ++i) {
            int row = tm * 16 + i + hi * 8;
            float v = epi_apply(EPI, acc[i] + bv);
            if (C)  C[(long)row * ldc + col] = v;
            if (Cb) Cb[(long)row * ldc + col] = f2bf(v);
        }
    }
}

// ---------------- elementwise / scatter kernels ----------------
__global__ void k_f2b(const float* __restrict__ in, __bf16* __restrict__ out, long n) {
    long i = (long)blockIdx.x * blockDim.x + threadIdx.x;
    if (i < n) out[i] = f2bf(in[i]);
}

__global__ void k_init(float* __restrict__ p, float v, long n) {
    long i = (long)blockIdx.x * blockDim.x + threadIdx.x;
    if (i < n) p[i] = v;
}

// W[Ks, NcSrc] (f32, row-major) -> Wt[NcDst, Kp] (bf16), zero-padded
__global__ void k_transpose_w(const float* __restrict__ W, __bf16* __restrict__ Wt,
                              int Ks, int Kp, int NcSrc, int NcDst) {
    long i = (long)blockIdx.x * blockDim.x + threadIdx.x;
    long tot = (long)NcDst * Kp;
    if (i >= tot) return;
    int n = (int)(i / Kp), k = (int)(i % Kp);
    float v = (k < Ks && n < NcSrc) ? W[(long)k * NcSrc + n] : 0.f;
    Wt[i] = f2bf(v);
}

// kv_in = [x[src] | eattr | 0-pad] -> bf16 [E,288];  xdst = x[dst] -> bf16 [E,256]
__global__ void k_gather(const float* __restrict__ x, const float* __restrict__ ea,
                         const int* __restrict__ src, const int* __restrict__ dst,
                         __bf16* __restrict__ kvb, __bf16* __restrict__ xd) {
    long i = (long)blockIdx.x * blockDim.x + threadIdx.x;
    if (i >= (long)EE * KVP) return;
    int e = (int)(i / KVP), c = (int)(i % KVP);
    int s = src[e];
    float v = (c < HH) ? x[(long)s * HH + c]
                       : ((c < HH + FF) ? ea[(long)e * FF + (c - HH)] : 0.f);
    kvb[i] = f2bf(v);
    if (c < HH) {
        int d = dst[e];
        xd[(long)e * HH + c] = f2bf(x[(long)d * HH + c]);
    }
}

__global__ void k_scores(const float* __restrict__ q, const float* __restrict__ k,
                         const float* __restrict__ prior, float* __restrict__ sc) {
    long i = (long)blockIdx.x * blockDim.x + threadIdx.x;
    if (i >= (long)EE * NHH) return;
    int e = (int)(i >> 3), h = (int)(i & 7);
    const float* qp = q + (long)e * HH + h * HD;
    const float* kp = k + (long)e * HH + h * HD;
    float s = 0.f;
#pragma unroll
    for (int d = 0; d < HD; ++d) s += qp[d] * kp[d];
    sc[i] = s * 0.17677669529663687f * prior[h];
}

__global__ void k_segmax(const float* __restrict__ sc, const int* __restrict__ dst,
                         float* __restrict__ m) {
    long i = (long)blockIdx.x * blockDim.x + threadIdx.x;
    if (i >= (long)EE * NHH) return;
    int e = (int)(i >> 3), h = (int)(i & 7);
    atomicMaxF(&m[(long)dst[e] * NHH + h], sc[i]);
}

__global__ void k_expden(float* __restrict__ sc, const int* __restrict__ dst,
                         const float* __restrict__ m, float* __restrict__ den) {
    long i = (long)blockIdx.x * blockDim.x + threadIdx.x;
    if (i >= (long)EE * NHH) return;
    int e = (int)(i >> 3), h = (int)(i & 7);
    float ex = expf(sc[i] - m[(long)dst[e] * NHH + h]);
    sc[i] = ex;
    atomicAdd(&den[(long)dst[e] * NHH + h], ex);
}

__global__ void k_agg(const float* __restrict__ sc, const float* __restrict__ den,
                      const float* __restrict__ v, const int* __restrict__ dst,
                      float* __restrict__ agg) {
    long i = (long)blockIdx.x * blockDim.x + threadIdx.x;
    if (i >= (long)EE * HH) return;
    int e = (int)(i >> 8), c = (int)(i & 255);
    int h = c >> 5;
    int d = dst[e];
    float w = sc[(long)e * NHH + h] / (den[(long)d * NHH + h] + 1e-16f);
    atomicAdd(&agg[(long)d * HH + c], w * v[i]);
}

__global__ void k_softmax6(const float* __restrict__ logits, float* __restrict__ iw) {
    long n = (long)blockIdx.x * blockDim.x + threadIdx.x;
    if (n >= NN) return;
    const float* l = logits + n * 16;
    float mx = l[0];
#pragma unroll
    for (int r = 1; r < RR; ++r) mx = fmaxf(mx, l[r]);
    float e[RR], s = 0.f;
#pragma unroll
    for (int r = 0; r < RR; ++r) { e[r] = expf(l[r] - mx); s += e[r]; }
    float inv = 1.f / s;
#pragma unroll
    for (int r = 0; r < RR; ++r) iw[n * RR + r] = e[r] * inv;
}

__global__ void k_interagg(const float* __restrict__ iw, const __bf16* __restrict__ rel,
                           __bf16* __restrict__ comb) {
    long i = (long)blockIdx.x * blockDim.x + threadIdx.x;
    if (i >= (long)NN * HH) return;
    int n = (int)(i >> 8), c = (int)(i & 255);
    float s = 0.f;
#pragma unroll
    for (int r = 0; r < RR; ++r)
        s += iw[(long)n * RR + r] * bf2f(rel[((long)r * NN + n) * HH + c]);
    comb[(long)n * (2 * HH) + c] = f2bf(s);
}

// meta-paths: p0 = rel2+rel3, p1 = rel4+rel0, p2 = rel1+rel5
__global__ void k_paths(const __bf16* __restrict__ rel, __bf16* __restrict__ paths) {
    long i = (long)blockIdx.x * blockDim.x + threadIdx.x;
    if (i >= (long)NN * HH) return;
    int n = (int)(i >> 8), c = (int)(i & 255);
    float r0 = bf2f(rel[((long)0 * NN + n) * HH + c]);
    float r1 = bf2f(rel[((long)1 * NN + n) * HH + c]);
    float r2 = bf2f(rel[((long)2 * NN + n) * HH + c]);
    float r3 = bf2f(rel[((long)3 * NN + n) * HH + c]);
    float r4 = bf2f(rel[((long)4 * NN + n) * HH + c]);
    float r5 = bf2f(rel[((long)5 * NN + n) * HH + c]);
    long b = (long)n * (3 * HH) + c;
    paths[b]          = f2bf(r2 + r3);
    paths[b + HH]     = f2bf(r4 + r0);
    paths[b + 2 * HH] = f2bf(r1 + r5);
}

__global__ void k_attn(const float* __restrict__ tb, const float* __restrict__ Wa2,
                       float* __restrict__ attn) {
    long n = (long)blockIdx.x * blockDim.x + threadIdx.x;
    if (n >= NN) return;
    float l[3];
#pragma unroll
    for (int p = 0; p < 3; ++p) {
        const float* t = tb + ((long)n * 3 + p) * 128;
        float s = 0.f;
        for (int d = 0; d < 128; ++d) s += t[d] * Wa2[d];
        l[p] = s;
    }
    float mx = fmaxf(l[0], fmaxf(l[1], l[2]));
    float e0 = expf(l[0] - mx), e1 = expf(l[1] - mx), e2 = expf(l[2] - mx);
    float inv = 1.f / (e0 + e1 + e2);
    attn[n * 3 + 0] = e0 * inv;
    attn[n * 3 + 1] = e1 * inv;
    attn[n * 3 + 2] = e2 * inv;
}

// meta = LayerNorm(sum_p attn * stacked) -> bf16 into combined[:, 256:512]
__global__ void k_meta(const float* __restrict__ attn, const float* __restrict__ st,
                       const float* __restrict__ g, const float* __restrict__ b,
                       __bf16* __restrict__ comb) {
    int n = blockIdx.x, t = threadIdx.x;
    __shared__ float red[HH];
    float a0 = attn[n * 3 + 0], a1 = attn[n * 3 + 1], a2 = attn[n * 3 + 2];
    const float* s = st + (long)n * (3 * HH);
    float v = a0 * s[t] + a1 * s[HH + t] + a2 * s[2 * HH + t];
    red[t] = v; __syncthreads();
    for (int o = HH / 2; o > 0; o >>= 1) { if (t < o) red[t] += red[t + o]; __syncthreads(); }
    float mu = red[0] * (1.f / HH); __syncthreads();
    float d = v - mu;
    red[t] = d * d; __syncthreads();
    for (int o = HH / 2; o > 0; o >>= 1) { if (t < o) red[t] += red[t + o]; __syncthreads(); }
    float var = red[0] * (1.f / HH);
    float y = d * rsqrtf(var + 1e-5f) * g[t] + b[t];
    comb[(long)n * (2 * HH) + HH + t] = f2bf(y);
}

// out = LayerNorm(x + combined)
__global__ void k_outln(const float* __restrict__ x, const float* __restrict__ cmb,
                        const float* __restrict__ g, const float* __restrict__ b,
                        float* __restrict__ out) {
    int n = blockIdx.x, t = threadIdx.x;
    __shared__ float red[HH];
    float v = x[(long)n * HH + t] + cmb[(long)n * HH + t];
    red[t] = v; __syncthreads();
    for (int o = HH / 2; o > 0; o >>= 1) { if (t < o) red[t] += red[t + o]; __syncthreads(); }
    float mu = red[0] * (1.f / HH); __syncthreads();
    float d = v - mu;
    red[t] = d * d; __syncthreads();
    for (int o = HH / 2; o > 0; o >>= 1) { if (t < o) red[t] += red[t + o]; __syncthreads(); }
    float var = red[0] * (1.f / HH);
    out[(long)n * HH + t] = d * rsqrtf(var + 1e-5f) * g[t] + b[t];
}

// ---------------- host ----------------
extern "C" void kernel_launch(void* const* d_in, const int* in_sizes, int n_in,
                              void* d_out, int out_size, void* d_ws, size_t ws_size,
                              hipStream_t stream) {
    const float* x     = (const float*)d_in[0];
    const int*   eidx  = (const int*)d_in[1];
    const float* eattr = (const float*)d_in[2];
    const float* Wq    = (const float*)d_in[3];
    const float* bq    = (const float*)d_in[4];
    const float* Wk    = (const float*)d_in[5];
    const float* bk    = (const float*)d_in[6];
    const float* Wv    = (const float*)d_in[7];
    const float* bv    = (const float*)d_in[8];
    const float* prior = (const float*)d_in[9];
    const float* Wm    = (const float*)d_in[10];
    const float* bm    = (const float*)d_in[11];
    const float* W_ir1 = (const float*)d_in[12];
    const float* b_ir1 = (const float*)d_in[13];
    const float* W_ir2 = (const float*)d_in[14];
    const float* b_ir2 = (const float*)d_in[15];
    const float* Wmp   = (const float*)d_in[16];
    const float* bmp   = (const float*)d_in[17];
    const float* Wa1   = (const float*)d_in[18];
    const float* ba1   = (const float*)d_in[19];
    const float* Wa2   = (const float*)d_in[20];
    const float* g_meta= (const float*)d_in[21];
    const float* b_meta= (const float*)d_in[22];
    const float* Wc    = (const float*)d_in[23];
    const float* bc    = (const float*)d_in[24];
    const float* g_out = (const float*)d_in[25];
    const float* b_out = (const float*)d_in[26];
    float* out = (float*)d_out;

    char* cur = (char*)d_ws;
    auto alloc = [&](size_t bytes) -> char* {
        char* p = cur; cur += ((bytes + 255) & ~(size_t)255); return p;
    };
    // persistent
    __bf16* xb     = (__bf16*)alloc((size_t)NN * HH * 2);
    __bf16* WqT    = (__bf16*)alloc((size_t)RR * HH * HH * 2);
    __bf16* WkT    = (__bf16*)alloc((size_t)RR * HH * KVP * 2);
    __bf16* WvT    = (__bf16*)alloc((size_t)RR * HH * KVP * 2);
    __bf16* WmT    = (__bf16*)alloc((size_t)RR * HH * HH * 2);
    __bf16* Wir1T  = (__bf16*)alloc((size_t)HH * (RR * HH) * 2);
    __bf16* Wir2T  = (__bf16*)alloc((size_t)16 * HH * 2);
    __bf16* WmpT   = (__bf16*)alloc((size_t)3 * HH * HH * 2);
    __bf16* Wa1T   = (__bf16*)alloc((size_t)128 * HH * 2);
    __bf16* WcT    = (__bf16*)alloc((size_t)HH * (2 * HH) * 2);
    __bf16* rel    = (__bf16*)alloc((size_t)RR * NN * HH * 2);
    __bf16* hb     = (__bf16*)alloc((size_t)NN * HH * 2);
    float*  logits = (float*)alloc((size_t)NN * 16 * 4);
    float*  interw = (float*)alloc((size_t)NN * RR * 4);
    __bf16* combined = (__bf16*)alloc((size_t)NN * 2 * HH * 2);
    __bf16* pathsb   = (__bf16*)alloc((size_t)NN * 3 * HH * 2);
    __bf16* stackedb = (__bf16*)alloc((size_t)NN * 3 * HH * 2);
    float*  attn   = (float*)alloc((size_t)NN * 3 * 4);
    float*  comb   = (float*)alloc((size_t)NN * HH * 4);
    // union region: per-relation scratch vs. stage-2 scratch
    char* uni = cur;
    char* c1 = uni;
    auto alloc1 = [&](size_t bytes) -> char* {
        char* p = c1; c1 += ((bytes + 255) & ~(size_t)255); return p;
    };
    __bf16* kvb = (__bf16*)alloc1((size_t)EE * KVP * 2);
    __bf16* xdb = (__bf16*)alloc1((size_t)EE * HH * 2);
    float*  qf  = (float*)alloc1((size_t)EE * HH * 4);
    float*  kf  = (float*)alloc1((size_t)EE * HH * 4);
    float*  vf  = (float*)alloc1((size_t)EE * HH * 4);
    float*  sc  = (float*)alloc1((size_t)EE * NHH * 4);
    float*  mmax= (float*)alloc1((size_t)NN * NHH * 4);
    float*  den = (float*)alloc1((size_t)NN * NHH * 4);
    float*  agg = (float*)alloc1((size_t)NN * HH * 4);
    __bf16* aggb= (__bf16*)alloc1((size_t)NN * HH * 2);
    char* c2 = uni;
    auto alloc2 = [&](size_t bytes) -> char* {
        char* p = c2; c2 += ((bytes + 255) & ~(size_t)255); return p;
    };
    float* stackedf = (float*)alloc2((size_t)NN * 3 * HH * 4);
    float* tanhbuf  = (float*)alloc2((size_t)NN * 3 * 128 * 4);

    auto blk = [](long n) { return dim3((unsigned)((n + 255) / 256)); };
    auto launch_big = [&](int epi, const __bf16* A, long lda, const __bf16* Bt, long ldb,
                          int K, const float* bias, float* C, __bf16* Cb, long ldc,
                          int M, int Nc, int nValid) {
        int tilesM = (M + 31) / 32, tilesN = Nc / 64;
        long waves = (long)tilesM * tilesN;
        dim3 grid((unsigned)((waves + 7) / 8));
        if (epi == 0)
            gemm_big<0><<<grid, 256, 0, stream>>>(A, lda, Bt, ldb, K, bias, C, Cb, ldc, M, tilesM, tilesN, nValid);
        else if (epi == 1)
            gemm_big<1><<<grid, 256, 0, stream>>>(A, lda, Bt, ldb, K, bias, C, Cb, ldc, M, tilesM, tilesN, nValid);
        else
            gemm_big<2><<<grid, 256, 0, stream>>>(A, lda, Bt, ldb, K, bias, C, Cb, ldc, M, tilesM, tilesN, nValid);
    };

    // ---- prep: convert x, transpose+convert weights ----
    k_f2b<<<blk((long)NN * HH), 256, 0, stream>>>(x, xb, (long)NN * HH);
    for (int r = 0; r < RR; ++r) {
        k_transpose_w<<<blk((long)HH * HH), 256, 0, stream>>>(Wq + (long)r * HH * HH, WqT + (long)r * HH * HH, HH, HH, HH, HH);
        k_transpose_w<<<blk((long)HH * KVP), 256, 0, stream>>>(Wk + (long)r * (HH + FF) * HH, WkT + (long)r * HH * KVP, HH + FF, KVP, HH, HH);
        k_transpose_w<<<blk((long)HH * KVP), 256, 0, stream>>>(Wv + (long)r * (HH + FF) * HH, WvT + (long)r * HH * KVP, HH + FF, KVP, HH, HH);
        k_transpose_w<<<blk((long)HH * HH), 256, 0, stream>>>(Wm + (long)r * HH * HH, WmT + (long)r * HH * HH, HH, HH, HH, HH);
    }
    k_transpose_w<<<blk((long)HH * RR * HH), 256, 0, stream>>>(W_ir1, Wir1T, RR * HH, RR * HH, HH, HH);
    k_transpose_w<<<blk((long)16 * HH), 256, 0, stream>>>(W_ir2, Wir2T, HH, HH, RR, 16);
    for (int p = 0; p < 3; ++p)
        k_transpose_w<<<blk((long)HH * HH), 256, 0, stream>>>(Wmp + (long)p * HH * HH, WmpT + (long)p * HH * HH, HH, HH, HH, HH);
    k_transpose_w<<<blk((long)128 * HH), 256, 0, stream>>>(Wa1, Wa1T, HH, HH, 128, 128);
    k_transpose_w<<<blk((long)HH * 2 * HH), 256, 0, stream>>>(Wc, WcT, 2 * HH, 2 * HH, HH, HH);

    // ---- per-relation attention ----
    for (int r = 0; r < RR; ++r) {
        const int* src = eidx + (long)r * 2 * EE;
        const int* dst = src + EE;
        k_gather<<<blk((long)EE * KVP), 256, 0, stream>>>(x, eattr + (long)r * EE * FF, src, dst, kvb, xdb);
        launch_big(0, xdb, HH,  WqT + (long)r * HH * HH,  HH,  HH,  bq + r * HH, qf, nullptr, HH, EE, HH, HH);
        launch_big(0, kvb, KVP, WkT + (long)r * HH * KVP, KVP, KVP, bk + r * HH, kf, nullptr, HH, EE, HH, HH);
        launch_big(0, kvb, KVP, WvT + (long)r * HH * KVP, KVP, KVP, bv + r * HH, vf, nullptr, HH, EE, HH, HH);
        k_scores<<<blk((long)EE * NHH), 256, 0, stream>>>(qf, kf, prior + r * NHH, sc);
        k_init<<<blk((long)NN * NHH), 256, 0, stream>>>(mmax, -3.0e38f, (long)NN * NHH);
        k_init<<<blk((long)NN * NHH), 256, 0, stream>>>(den, 0.f, (long)NN * NHH);
        k_segmax<<<blk((long)EE * NHH), 256, 0, stream>>>(sc, dst, mmax);
        k_expden<<<blk((long)EE * NHH), 256, 0, stream>>>(sc, dst, mmax, den);
        k_init<<<blk((long)NN * HH), 256, 0, stream>>>(agg, 0.f, (long)NN * HH);
        k_agg<<<blk((long)EE * HH), 256, 0, stream>>>(sc, den, vf, dst, agg);
        k_f2b<<<blk((long)NN * HH), 256, 0, stream>>>(agg, aggb, (long)NN * HH);
        launch_big(1, aggb, HH, WmT + (long)r * HH * HH, HH, HH, bm + r * HH,
                   nullptr, rel + (long)r * NN * HH, HH, NN, HH, HH);
    }

    // ---- inter-relation mixing ----
    {
        int tilesM = (NN + 31) / 32, tilesN = HH / 64;
        long waves = (long)tilesM * tilesN;
        dim3 grid((unsigned)((waves + 7) / 8));
        gemm_relcat_big<1><<<grid, 256, 0, stream>>>(rel, Wir1T, b_ir1, nullptr, hb, HH,
                                                     tilesM, tilesN, HH);
    }
    {   // logits: Nc = 16 (only 6 valid) -> small kernel
        long waves = (long)(NN / 16) * 1;
        dim3 grid((unsigned)((waves + 7) / 8));
        gemm_bf16<0><<<grid, 256, 0, stream>>>(hb, HH, Wir2T, HH, HH, b_ir2,
                                               logits, nullptr, 16, NN / 16, 1, RR);
    }
    k_softmax6<<<blk(NN), 256, 0, stream>>>(logits, interw);
    k_interagg<<<blk((long)NN * HH), 256, 0, stream>>>(interw, rel, combined);

    // ---- meta-path attention ----
    k_paths<<<blk((long)NN * HH), 256, 0, stream>>>(rel, pathsb);
    for (int p = 0; p < 3; ++p)
        launch_big(0, pathsb + p * HH, 3 * HH, WmpT + (long)p * HH * HH, HH, HH, bmp + p * HH,
                   stackedf + p * HH, stackedb + p * HH, 3 * HH, NN, HH, HH);
    launch_big(2, stackedb, HH, Wa1T, HH, HH, ba1, tanhbuf, nullptr, 128,
               3 * NN, 128, 128);
    k_attn<<<blk(NN), 256, 0, stream>>>(tanhbuf, Wa2, attn);
    k_meta<<<dim3(NN), 256, 0, stream>>>(attn, stackedf, g_meta, b_meta, combined);

    // ---- combine + output LN ----
    launch_big(1, combined, 2 * HH, WcT, 2 * HH, 2 * HH, bc, comb, nullptr, HH,
               NN, HH, HH);
    k_outln<<<dim3(NN), 256, 0, stream>>>(x, comb, g_out, b_out, out);
}
